// JumpFlow_39530878992479
// MI455X (gfx1250) — matrix operand order repaired and checked by reading, not compiled, not run on hardware
//
#include <hip/hip_runtime.h>
#include <math.h>

// ---------------- problem constants (from reference) ----------------
#define B_    256
#define T_    128
#define HID_  64
#define MC_   30
#define L_    2
#define NPATH (B_ * MC_)        // 7680 MC rows
#define NROWS (NPATH + B_)      // 7936 rows per step (paths + h2)
#define XS    66                // f32 x-tile row stride (bank-conflict pad)
#define ZS    72                // bf16 staging row stride (16B-aligned pad)

typedef __attribute__((ext_vector_type(16))) __bf16 v16bf;
typedef __attribute__((ext_vector_type(8)))  __bf16 v8bf;
typedef __attribute__((ext_vector_type(8)))  float  v8f;

__device__ __forceinline__ v8f wmma_bf16(v16bf a, v16bf b, v8f c) {
  // D = A(16x32 bf16) * B(32x16 bf16) + C(16x16 f32)
  return __builtin_amdgcn_wmma_f32_16x16x32_bf16(
      /*neg_a=*/false, a, /*neg_b=*/false, b,
      /*c_mod=*/(short)0, c, /*reuse_a=*/false, /*reuse_b=*/false);
}

// Hardware tanh on gfx1250 (V_TANH_F32) with branch-free fallback.
__device__ __forceinline__ float fast_tanh(float x) {
#if __has_builtin(__builtin_amdgcn_tanhf)
  return __builtin_amdgcn_tanhf(x);
#elif __has_builtin(__builtin_amdgcn_tanh_f32)
  return __builtin_amdgcn_tanh_f32(x);
#else
  float e = __expf(-2.0f * __builtin_fabsf(x));   // v_exp_f32
  float r = (1.0f - e) / (1.0f + e);              // v_rcp_f32 path, no branches
  return __builtin_copysignf(r, x);
#endif
}

__device__ __forceinline__ float sigm(float x) { return 1.0f / (1.0f + __expf(-x)); }

// Branch-free softplus: select lowers to v_cndmask (no exec-mask divergence).
__device__ __forceinline__ float softplusf(float x) {
  float sp = __logf(1.0f + __expf(x));
  return (x > 20.0f) ? x : sp;
}

// A-fragment (16x32 bf16) from an LDS tile with row stride ZS.
// ISA 7.12.2: lane L -> M = L&15; elems 0..7 = K = kc*32 + (L>>4)*8 + i,
// elems 8..15 = K = kc*32 + 16 + (L>>4)*8 + i.
__device__ __forceinline__ v16bf load_a(const __bf16* tile, int lane, int kc) {
  int m  = lane & 15;
  int kg = (lane >> 4) << 3;
  const __bf16* p = tile + m * ZS + kc * 32 + kg;
  v8bf lo = *(const v8bf*)(p);
  v8bf hi = *(const v8bf*)(p + 16);
  v16bf a;
#pragma unroll
  for (int i = 0; i < 8; ++i) { a[i] = lo[i]; a[i + 8] = hi[i]; }
  return a;
}

// B-fragment (32x16 bf16) from a row-major [K][N] bf16 weight matrix.
// ISA 7.12.2 (B layout): lane L -> N = n0 + (L&15); elems 0..7 = K = (L>>4)*8 + i,
// elems 8..15 = K = 16 + (L>>4)*8 + i.
// `p` already points at W + k0*ncols + n0.
__device__ __forceinline__ v16bf load_b(const __bf16* p, int ncols, int lane) {
  int n  = lane & 15;
  int kg = (lane >> 4) << 3;
  v16bf b;
#pragma unroll
  for (int i = 0; i < 8; ++i) {
    b[i]     = p[(kg + i) * ncols + n];
    b[i + 8] = p[(16 + kg + i) * ncols + n];
  }
  return b;
}

// ---------------- per-step flow kernel (WMMA heavy) ----------------
// 62 blocks x 256 threads; each wave owns a 16-row tile of the 7936 rows.
__global__ __launch_bounds__(256) void flow_step_kernel(
    const float* __restrict__ times,   // (B,T,1)
    const float* __restrict__ u,       // (T,MC,1)
    const float* __restrict__ h,       // (B,HID) carry
    const __bf16* __restrict__ W0bf,   // [L][64][64]
    const __bf16* __restrict__ W1bf,   // [L][64][64]
    const __bf16* __restrict__ W2bf,   // [L][64][128]
    const float* __restrict__ w0t,     // [L][64] (row 65 of fW0, the t column)
    const float* __restrict__ fb0, const float* __restrict__ fb1,
    const float* __restrict__ fb2, const float* __restrict__ ftw,
    const float* __restrict__ Wi, const float* __restrict__ bi,
    float* __restrict__ hidden_out,    // (B,T,HID) = d_out+1
    float* __restrict__ sp_buf,        // (B*MC) softplus of path intensities
    float* __restrict__ lam_out,       // (B) intensity at h2
    int t_idx)
{
  __shared__ float sX[8][16][XS];                 // f32 state tile per wave
  __shared__ __align__(16) __bf16 sZ[8][16][ZS];  // bf16 matmul-input staging
  __shared__ float sT[8][16];                     // per-row time scalar

  const int tid  = threadIdx.x;
  const int w    = tid >> 5;
  const int lane = tid & 31;
  const int kgrp = lane >> 4;      // 0/1 lane half-group
  const int ln   = lane & 15;
  const int r0   = blockIdx.x * 128 + w * 16;

  // ---- load x rows (x = h[b] broadcast) and per-row t ----
#pragma unroll
  for (int i = 0; i < 32; ++i) {
    int idx = lane + i * 32;            // 0..1023
    int row = idx >> 6, col = idx & 63;
    int r = r0 + row;
    int b = (r < NPATH) ? (r / MC_) : (r - NPATH);
    sX[w][row][col] = h[b * HID_ + col];
  }
  if (lane < 16) {
    int r = r0 + lane;
    float tv;
    if (r < NPATH) {
      int b = r / MC_, j = r % MC_;
      tv = u[t_idx * MC_ + j] * times[b * T_ + t_idx];
    } else {
      tv = times[(r - NPATH) * T_ + t_idx];
    }
    sT[w][lane] = tv;
  }
  __syncthreads();

  float tm[8];                        // t for the 8 C-fragment rows of this lane
#pragma unroll
  for (int v = 0; v < 8; ++v) tm[v] = sT[w][kgrp * 8 + v];

#pragma unroll
  for (int l = 0; l < L_; ++l) {
    // ---- stage z = x * m (mask halves by layer parity), f32 -> bf16 ----
#pragma unroll
    for (int i = 0; i < 32; ++i) {
      int idx = lane + i * 32;
      int row = idx >> 6, col = idx & 63;
      bool on = (col < 32) == (l == 0);
      sZ[w][row][col] = (__bf16)(on ? sX[w][row][col] : 0.0f);
    }
    __syncthreads();

    // ---- GEMM1: a = tanh([z,t] @ W0 + b0), K=64 via WMMA + t rank-1 in VALU
    v16bf fa0 = load_a(&sZ[w][0][0], lane, 0);
    v16bf fa1 = load_a(&sZ[w][0][0], lane, 1);
    const __bf16* W0l = W0bf + l * HID_ * HID_;
#pragma unroll
    for (int nt = 0; nt < 4; ++nt) {
      int n0 = nt * 16, nc = n0 + ln;
      v8f acc = {0.f, 0.f, 0.f, 0.f, 0.f, 0.f, 0.f, 0.f};
      acc = wmma_bf16(fa0, load_b(W0l + 0 * HID_ + n0, HID_, lane), acc);
      acc = wmma_bf16(fa1, load_b(W0l + 32 * HID_ + n0, HID_, lane), acc);
      float wt = w0t[l * HID_ + nc];
      float bb = fb0[l * HID_ + nc];
#pragma unroll
      for (int v = 0; v < 8; ++v) {
        float av = fast_tanh(acc[v] + tm[v] * wt + bb);
        sZ[w][kgrp * 8 + v][nc] = (__bf16)av;   // C-layout -> row-major restage
      }
    }
    __syncthreads();

    // ---- GEMM2: a = tanh(a @ W1 + b1) ----
    fa0 = load_a(&sZ[w][0][0], lane, 0);
    fa1 = load_a(&sZ[w][0][0], lane, 1);
    const __bf16* W1l = W1bf + l * HID_ * HID_;
#pragma unroll
    for (int nt = 0; nt < 4; ++nt) {
      int n0 = nt * 16, nc = n0 + ln;
      v8f acc = {0.f, 0.f, 0.f, 0.f, 0.f, 0.f, 0.f, 0.f};
      acc = wmma_bf16(fa0, load_b(W1l + 0 * HID_ + n0, HID_, lane), acc);
      acc = wmma_bf16(fa1, load_b(W1l + 32 * HID_ + n0, HID_, lane), acc);
      float bb = fb1[l * HID_ + nc];
#pragma unroll
      for (int v = 0; v < 8; ++v) {
        float av = fast_tanh(acc[v] + bb);
        sZ[w][kgrp * 8 + v][nc] = (__bf16)av;
      }
    }
    __syncthreads();

    // ---- GEMM3: ss = a @ W2 + b2; coupling update on x, in C-layout ----
    fa0 = load_a(&sZ[w][0][0], lane, 0);
    fa1 = load_a(&sZ[w][0][0], lane, 1);
    const __bf16* W2l = W2bf + l * HID_ * 128;
#pragma unroll
    for (int nt = 0; nt < 4; ++nt) {
      int n0 = nt * 16;
      int c  = n0 + ln;
      v8f accS = {0.f, 0.f, 0.f, 0.f, 0.f, 0.f, 0.f, 0.f};   // scale cols c
      v8f accH = {0.f, 0.f, 0.f, 0.f, 0.f, 0.f, 0.f, 0.f};   // shift cols 64+c
      accS = wmma_bf16(fa0, load_b(W2l + 0 * 128 + n0, 128, lane), accS);
      accS = wmma_bf16(fa1, load_b(W2l + 32 * 128 + n0, 128, lane), accS);
      accH = wmma_bf16(fa0, load_b(W2l + 0 * 128 + 64 + n0, 128, lane), accH);
      accH = wmma_bf16(fa1, load_b(W2l + 32 * 128 + 64 + n0, 128, lane), accH);
      float mv  = ((c < 32) == (l == 0)) ? 1.0f : 0.0f;
      float fsc = ftw[l * 128 + c];
      float fsh = ftw[l * 128 + 64 + c];
      float b2s = fb2[l * 128 + c];
      float b2h = fb2[l * 128 + 64 + c];
#pragma unroll
      for (int v = 0; v < 8; ++v) {
        int   M  = kgrp * 8 + v;
        float xv = sX[w][M][c];
        float t  = tm[v];
        float sc = accS[v] + b2s;
        float sh = accH[v] + b2h;
        float xn = mv * xv +
                   (1.0f - mv) * (xv * __expf(sc * fast_tanh(t * fsc)) +
                                  sh * fast_tanh(t * fsh));
        sX[w][M][c] = xn;   // element touched only by this lane -> race-free
      }
    }
    __syncthreads();
  }

  // ---- intensity: softplus(x @ Wi + bi), outputs ----
  if (lane < 16) {
    float dot = bi[0];
#pragma unroll
    for (int c2 = 0; c2 < HID_; ++c2) dot += sX[w][lane][c2] * Wi[c2];
    float sp = softplusf(dot);
    int r = r0 + lane;
    if (r < NPATH) sp_buf[r] = sp;
    else           lam_out[r - NPATH] = sp;
  }
  if (r0 >= NPATH) {  // h2 rows: emit hidden state for this step
#pragma unroll
    for (int i = 0; i < 32; ++i) {
      int idx = lane + i * 32;
      int row = idx >> 6, col = idx & 63;
      int b = r0 + row - NPATH;
      hidden_out[(b * T_ + t_idx) * HID_ + col] = sX[w][row][col];
    }
  }
}

// ---------------- per-step LSTM cell + nll ----------------
__global__ __launch_bounds__(256) void lstm_step_kernel(
    const float* __restrict__ times, const float* __restrict__ gmark,
    const float* __restrict__ W_ih,  const float* __restrict__ W_hh,
    const float* __restrict__ hidden, const float* __restrict__ lam,
    const float* __restrict__ sp_buf,
    float* __restrict__ h, float* __restrict__ c, float* __restrict__ nll,
    int t_idx)
{
  int b = blockIdx.x, g = threadIdx.x;
  __shared__ float sh2[HID_];
  __shared__ float sg[256];
  float tb = times[b * T_ + t_idx];
  if (g < HID_) sh2[g] = hidden[(b * T_ + t_idx) * HID_ + g];
  __syncthreads();

  float acc = gmark[b * 256 + g] + tb * W_ih[g * 65];  // + t * W_ih[:,0]
  const float* whr = W_hh + g * HID_;
#pragma unroll 8
  for (int k = 0; k < HID_; ++k) acc += whr[k] * sh2[k];
  sg[g] = acc;
  __syncthreads();

  if (g < HID_) {
    float ig = sg[g], fg = sg[64 + g], gg = sg[128 + g], og = sg[192 + g];
    float cp = c[b * HID_ + g];
    float cn = sigm(fg) * cp + sigm(ig) * fast_tanh(gg);
    c[b * HID_ + g] = cn;
    h[b * HID_ + g] = sigm(og) * fast_tanh(cn);
  }
  if (g == 0) {   // deterministic MC mean (no atomics)
    float s = 0.0f;
    for (int j = 0; j < MC_; ++j) s += sp_buf[b * MC_ + j];
    float integral = s * (1.0f / (float)MC_) * tb;
    nll[b * T_ + t_idx] = -__logf(lam[b]) + integral;
  }
}

// ---------------- prep / init / loss ----------------
__global__ __launch_bounds__(256) void prep_weights(
    const float* __restrict__ fW0, const float* __restrict__ fW1,
    const float* __restrict__ fW2,
    __bf16* __restrict__ W0bf, __bf16* __restrict__ W1bf,
    __bf16* __restrict__ W2bf, float* __restrict__ w0t)
{
  int i = blockIdx.x * blockDim.x + threadIdx.x;   // 0..16383
  if (i < L_ * 64 * 64) {
    int l = i >> 12, r = i & 4095;                 // fW0 is (L,65,64): first 64 rows
    W0bf[i] = (__bf16)fW0[l * 65 * 64 + r];
    W1bf[i] = (__bf16)fW1[i];
  }
  if (i < L_ * 64 * 128) W2bf[i] = (__bf16)fW2[i];
  if (i < L_ * 64) w0t[i] = fW0[(i >> 6) * 65 * 64 + 64 * 64 + (i & 63)];
}

__global__ __launch_bounds__(256) void prep_gmark(
    const int* __restrict__ marks, const float* __restrict__ emb,
    const float* __restrict__ W_ih, const float* __restrict__ b_ih,
    const float* __restrict__ b_hh, float* __restrict__ gmark)
{
  int b = blockIdx.x, g = threadIdx.x;
  __shared__ float se[HID_];
  int mk = marks[b * T_ + 1];                      // marks_e[:,1]
  if (g < HID_) se[g] = emb[mk * HID_ + g];
  __syncthreads();
  float acc = b_ih[g] + b_hh[g];
  const float* wr = W_ih + g * 65 + 1;
#pragma unroll 8
  for (int k = 0; k < HID_; ++k) acc += wr[k] * se[k];
  gmark[b * 256 + g] = acc;
}

__global__ __launch_bounds__(256) void init_state(float* h, float* c) {
  int i = blockIdx.x * blockDim.x + threadIdx.x;
  if (i < B_ * HID_) { h[i] = 0.0f; c[i] = 0.0f; }
}

__global__ __launch_bounds__(256) void loss_kernel(
    const float* __restrict__ nll, const float* __restrict__ mask,
    float* __restrict__ out)
{
  __shared__ float s1[256], s2[256];
  int tid = threadIdx.x;
  float a = 0.0f, m = 0.0f;
  for (int i = tid; i < B_ * T_; i += 256) {
    a += nll[i] * mask[i];
    m += mask[i];
  }
  s1[tid] = a; s2[tid] = m;
  __syncthreads();
  for (int s = 128; s > 0; s >>= 1) {
    if (tid < s) { s1[tid] += s1[tid + s]; s2[tid] += s2[tid + s]; }
    __syncthreads();
  }
  if (tid == 0) out[0] = s1[0] / s2[0];
}

// ---------------- host entry ----------------
extern "C" void kernel_launch(void* const* d_in, const int* in_sizes, int n_in,
                              void* d_out, int out_size, void* d_ws, size_t ws_size,
                              hipStream_t stream) {
  (void)in_sizes; (void)n_in; (void)out_size; (void)ws_size;
  const float* times = (const float*)d_in[0];
  const int*   marks = (const int*)d_in[1];
  const float* mask  = (const float*)d_in[2];
  const float* u     = (const float*)d_in[3];
  const float* emb   = (const float*)d_in[4];
  const float* fW0   = (const float*)d_in[5];
  const float* fb0   = (const float*)d_in[6];
  const float* fW1   = (const float*)d_in[7];
  const float* fb1   = (const float*)d_in[8];
  const float* fW2   = (const float*)d_in[9];
  const float* fb2   = (const float*)d_in[10];
  const float* ftw   = (const float*)d_in[11];
  const float* Wi    = (const float*)d_in[12];
  const float* bi    = (const float*)d_in[13];
  const float* W_ih  = (const float*)d_in[14];
  const float* W_hh  = (const float*)d_in[15];
  const float* b_ih  = (const float*)d_in[16];
  const float* b_hh  = (const float*)d_in[17];

  float* out    = (float*)d_out;            // out[0]=loss, out[1..]=hidden (B,T,HID)
  float* hidden = out + 1;

  // workspace carve-up (floats, then bf16)
  float* ws    = (float*)d_ws;
  float* h     = ws;                 // 16384
  float* cst   = h + B_ * HID_;      // 16384
  float* lam   = cst + B_ * HID_;    // 256
  float* nll   = lam + B_;           // 32768
  float* gmark = nll + B_ * T_;      // 65536
  float* spb   = gmark + B_ * 256;   // 7680
  float* w0t   = spb + NPATH;        // 128
  __bf16* W0bf = (__bf16*)(w0t + L_ * HID_);   // 8192 halves
  __bf16* W1bf = W0bf + L_ * HID_ * HID_;      // 8192
  __bf16* W2bf = W1bf + L_ * HID_ * HID_;      // 16384

  init_state<<<(B_ * HID_ + 255) / 256, 256, 0, stream>>>(h, cst);
  prep_weights<<<(L_ * 64 * 128 + 255) / 256, 256, 0, stream>>>(
      fW0, fW1, fW2, W0bf, W1bf, W2bf, w0t);
  prep_gmark<<<B_, 256, 0, stream>>>(marks, emb, W_ih, b_ih, b_hh, gmark);

  for (int t = 0; t < T_; ++t) {
    flow_step_kernel<<<NROWS / 128, 256, 0, stream>>>(
        times, u, h, W0bf, W1bf, W2bf, w0t, fb0, fb1, fb2, ftw, Wi, bi,
        hidden, spb, lam, t);
    lstm_step_kernel<<<B_, 256, 0, stream>>>(
        times, gmark, W_ih, W_hh, hidden, lam, spb, h, cst, nll, t);
  }
  loss_kernel<<<1, 256, 0, stream>>>(nll, mask, out);
}
